// WindowedAttention_19662360281649
// MI455X (gfx1250) — matrix operand order repaired and checked by reading, not compiled, not run on hardware
//
#include <hip/hip_runtime.h>
#include <hip/hip_bf16.h>
#include <math.h>

#define D_MODEL  1024
#define N_HEADS  16
#define HEAD_DIM 64
#define WINDOW   64
#define BATCH    2
#define SEQ      2048

typedef __bf16 bf16_t;
typedef __attribute__((ext_vector_type(16))) __bf16 v16bf;
typedef __attribute__((ext_vector_type(8)))  __bf16 v8bf;
typedef __attribute__((ext_vector_type(4)))  __bf16 v4bf;
typedef __attribute__((ext_vector_type(8)))  float  v8f;
typedef __attribute__((ext_vector_type(4)))  int    v4i;

#if defined(__has_builtin)
#if __has_builtin(__builtin_amdgcn_global_load_async_to_lds_b128)
#define HAVE_ASYNC_LDS 1
#endif
#if __has_builtin(__builtin_amdgcn_permlane16)
#define HAVE_PERMLANE 1
#endif
#endif

__device__ __forceinline__ v8f wmma_bf(v16bf a, v16bf b, v8f c) {
  // D = A(16x32 bf16) * B(32x16 bf16) + C(16x16 f32)
  return __builtin_amdgcn_wmma_f32_16x16x32_bf16(false, a, false, b, (short)0, c,
                                                 false, false);
}

// ---- 16-lane xor shuffles: v_permlane16_b32 (VALU) when available ----------
__device__ __forceinline__ float lanexor16(float v, int m) {
#if defined(HAVE_PERMLANE)
  unsigned lo, hi;
  switch (m) {  // sel[j] = j ^ m, nibble-packed (folds under unrolling)
    case 1:  lo = 0x67452301u; hi = 0xEFCDAB89u; break;
    case 2:  lo = 0x54761032u; hi = 0xDCFE98BAu; break;
    case 4:  lo = 0x32107654u; hi = 0xBA98FEDCu; break;
    default: lo = 0xFEDCBA98u; hi = 0x76543210u; break;
  }
  unsigned u = __builtin_bit_cast(unsigned, v);
  u = __builtin_amdgcn_permlane16(u, u, lo, hi, false, false);
  return __builtin_bit_cast(float, u);
#else
  return __shfl_xor(v, m, 32);
#endif
}
__device__ __forceinline__ float rowmax16(float v) {
  #pragma unroll
  for (int m = 1; m < 16; m <<= 1) v = fmaxf(v, lanexor16(v, m));
  return v;
}
__device__ __forceinline__ float rowsum16(float v) {
  #pragma unroll
  for (int m = 1; m < 16; m <<= 1) v += lanexor16(v, m);
  return v;
}

__device__ __forceinline__ void wait_async_lds() {
#if defined(HAVE_ASYNC_LDS)
#if __has_builtin(__builtin_amdgcn_s_wait_asynccnt)
  __builtin_amdgcn_s_wait_asynccnt(0);
#else
  asm volatile("s_wait_asynccnt 0x0" ::: "memory");
#endif
#endif
}

// ---------------------------------------------------------------- convert
__global__ void cvt_f32_bf16(const float* __restrict__ src,
                             bf16_t* __restrict__ dst, int n) {
  int i = (blockIdx.x * blockDim.x + threadIdx.x) * 4;
  int stride = gridDim.x * blockDim.x * 4;
  for (; i < n; i += stride) {
    float4 f = *(const float4*)(src + i);
    v4bf r = {(bf16_t)f.x, (bf16_t)f.y, (bf16_t)f.z, (bf16_t)f.w};
    *(v4bf*)(dst + i) = r;
  }
}

// ---------------------------------------------------------------- GEMM Y = X * W^T (+bias)
// X: M x K bf16 row-major (K <= 1024); W: N x K bf16 row-major.
// All 8 waves of a block share one 16-row M-tile: the X panel (16*K bf16,
// contiguous in memory) is staged once into LDS via async-to-LDS copies, then
// A fragments come from ds_load_b128. Each wave owns a 16x64 output tile.
template <bool F32OUT>
__global__ void __launch_bounds__(256) gemm_xwt(const bf16_t* __restrict__ X,
                                                const bf16_t* __restrict__ W,
                                                const float* __restrict__ bias,
                                                void* __restrict__ Y,
                                                int M, int N, int K) {
  __shared__ bf16_t xpanel[16 * 1024];   // 32 KB
  const int lane = threadIdx.x & 31;
  const int wib  = threadIdx.x >> 5;
  const int ntg  = N >> 6;               // 64-wide N groups
  const int bpm  = ntg >> 3;             // blocks per M-tile (8 waves/block)
  const int mt   = blockIdx.x / bpm;
  const int ng   = (blockIdx.x % bpm) * 8 + wib;
  const int lm = lane & 15;
  const int hw = lane >> 4;

  // ---- stage X panel: rows [mt*16, mt*16+16) are one contiguous 16*K block
  {
    const bf16_t* xsrc = X + (size_t)mt * 16 * K;
    const int iters = (16 * K) / (256 * 8);      // 8 when K=1024
    for (int i = 0; i < iters; ++i) {
      size_t off = (size_t)i * 2048 + (size_t)threadIdx.x * 8;  // halves
#if defined(HAVE_ASYNC_LDS)
      __builtin_amdgcn_global_load_async_to_lds_b128(
          (__attribute__((address_space(1))) v4i*)(xsrc + off),
          (__attribute__((address_space(3))) v4i*)(xpanel + off), 0, 0);
#else
      v8bf t = *(const v8bf*)(xsrc + off);
      *(v8bf*)(xpanel + off) = t;
#endif
    }
    wait_async_lds();
  }
  __syncthreads();

  v8f acc[4] = {};
  for (int kk = 0; kk < K; kk += 32) {
    if ((kk & 255) == 0) {
      #pragma unroll
      for (int nt = 0; nt < 4; ++nt) {   // stream W rows ahead (global_prefetch_b8)
        int n = ng * 64 + nt * 16 + lm;
        __builtin_prefetch(W + (size_t)n * K + kk + 256, 0, 0);
      }
    }
    // A fragment from LDS: lane = row lm; K(e) = 16*hw + e
    v8bf alo = *(const v8bf*)(xpanel + lm * K + kk + 16 * hw);
    v8bf ahi = *(const v8bf*)(xpanel + lm * K + kk + 16 * hw + 8);
    v16bf a;
    #pragma unroll
    for (int e = 0; e < 8; ++e) { a[e] = alo[e]; a[8 + e] = ahi[e]; }
    #pragma unroll
    for (int nt = 0; nt < 4; ++nt) {
      int n = ng * 64 + nt * 16 + lm;    // B lane = column n; k = 16*hw + e
      v16bf bfrag = *(const v16bf*)(W + (size_t)n * K + kk + 16 * hw);
      acc[nt] = wmma_bf(a, bfrag, acc[nt]);
    }
  }

  #pragma unroll
  for (int nt = 0; nt < 4; ++nt) {
    int n = ng * 64 + nt * 16 + lm;
    float bb = bias ? bias[n] : 0.f;
    #pragma unroll
    for (int r = 0; r < 8; ++r) {
      int m = mt * 16 + r + 8 * hw;      // C/D: row r+8*hw, col lm
      float v = acc[nt][r] + bb;
      if (F32OUT) ((float*)Y)[(size_t)m * N + n] = v;
      else        ((bf16_t*)Y)[(size_t)m * N + n] = (bf16_t)v;
    }
  }
}

// ---------------------------------------------------------------- windowed attention
// One wave = one 16-query tile of one (b,h). 5 fixed chunks of 32 keys covering
// [q0-64, q0+96) with masking -> uniform trip count across all waves.
__global__ void __launch_bounds__(256) attn_window(const bf16_t* __restrict__ Q,
                                                   const bf16_t* __restrict__ Km,
                                                   const bf16_t* __restrict__ V,
                                                   bf16_t* __restrict__ C) {
  __shared__ bf16_t smem[8 * 2560];   // per wave: VT[64][32] (2048) + P[16][32] (512)
  const int lane = threadIdx.x & 31;
  const int wib  = threadIdx.x >> 5;
  const int wid  = blockIdx.x * 8 + wib;
  const int qt = wid & 127;
  const int h  = (wid >> 7) & 15;
  const int b  = wid >> 11;
  const int q0 = qt * 16;
  const int lm = lane & 15;
  const int hw = lane >> 4;
  bf16_t* VT = smem + wib * 2560;
  bf16_t* PT = VT + 2048;

  // Q fragments (two K-steps over HEAD_DIM=64); A rows = lm for both halves
  const bf16_t* qrow = Q + ((size_t)(b * SEQ + q0 + lm) * D_MODEL) + h * HEAD_DIM;
  v16bf aq[2];
  #pragma unroll
  for (int kb = 0; kb < 2; ++kb) {
    v8bf lo = *(const v8bf*)(qrow + kb * 32 + 16 * hw);
    v8bf hi = *(const v8bf*)(qrow + kb * 32 + 16 * hw + 8);
    #pragma unroll
    for (int e = 0; e < 8; ++e) { aq[kb][e] = lo[e]; aq[kb][8 + e] = hi[e]; }
  }

  float m_run[8], l_run[8];
  v8f o[4] = {};
  #pragma unroll
  for (int r = 0; r < 8; ++r) { m_run[r] = -1e30f; l_run[r] = 0.f; }
  const float scale = 0.125f;   // 1/sqrt(64)

  for (int c = 0; c < 5; ++c) {
    const int kbase = q0 - 64 + c * 32;

    // stage V^T chunk: VT[hd][key_local], clamp OOB rows (their P is forced 0)
    {
      int j  = kbase + lane;
      int jc = j < 0 ? 0 : (j >= SEQ ? SEQ - 1 : j);
      const bf16_t* vrow = V + ((size_t)(b * SEQ + jc) * D_MODEL) + h * HEAD_DIM;
      #pragma unroll
      for (int c8 = 0; c8 < 8; ++c8) {
        v8bf vv = *(const v8bf*)(vrow + c8 * 8);
        #pragma unroll
        for (int u = 0; u < 8; ++u) VT[(c8 * 8 + u) * 32 + lane] = vv[u];
      }
    }
    __syncthreads();

    // scores S = Q * K^T for two 16-key subtiles
    v8f sacc[2];
    #pragma unroll
    for (int t = 0; t < 2; ++t) {
      sacc[t] = (v8f){};
      int j  = kbase + 16 * t + lm;
      int jc = j < 0 ? 0 : (j >= SEQ ? SEQ - 1 : j);
      const bf16_t* krow = Km + ((size_t)(b * SEQ + jc) * D_MODEL) + h * HEAD_DIM;
      #pragma unroll
      for (int kb = 0; kb < 2; ++kb) {
        v16bf bk = *(const v16bf*)(krow + kb * 32 + 16 * hw);
        sacc[t] = wmma_bf(aq[kb], bk, sacc[t]);
      }
    }

    // online softmax (row m = r + 8*hw lives across a 16-lane half-group)
    #pragma unroll
    for (int r = 0; r < 8; ++r) {
      int i  = q0 + r + 8 * hw;
      int j0 = kbase + lm, j1 = kbase + 16 + lm;
      bool ok0 = (j0 >= 0) && (j0 < SEQ) && (i - j0 <= WINDOW) && (j0 - i <= WINDOW);
      bool ok1 = (j1 >= 0) && (j1 < SEQ) && (i - j1 <= WINDOW) && (j1 - i <= WINDOW);
      float s0 = ok0 ? sacc[0][r] * scale : -1e30f;
      float s1 = ok1 ? sacc[1][r] * scale : -1e30f;
      float rm = rowmax16(fmaxf(s0, s1));
      float mn = fmaxf(m_run[r], rm);
      float al = __expf(m_run[r] - mn);
      m_run[r] = mn;
      float p0 = ok0 ? __expf(s0 - mn) : 0.f;   // force 0 so fully-masked chunks are exact
      float p1 = ok1 ? __expf(s1 - mn) : 0.f;
      float rs = rowsum16(p0 + p1);
      l_run[r] = l_run[r] * al + rs;
      #pragma unroll
      for (int nt = 0; nt < 4; ++nt) o[nt][r] *= al;
      PT[(r + 8 * hw) * 32 + lm]      = (bf16_t)p0;
      PT[(r + 8 * hw) * 32 + 16 + lm] = (bf16_t)p1;
    }
    __syncthreads();

    // O += P(16x32) * Vchunk(32x64): A from PT, B columns from VT rows
    v16bf ap;
    {
      v8bf lo = *(const v8bf*)(PT + lm * 32 + 16 * hw);
      v8bf hi = *(const v8bf*)(PT + lm * 32 + 16 * hw + 8);
      #pragma unroll
      for (int e = 0; e < 8; ++e) { ap[e] = lo[e]; ap[8 + e] = hi[e]; }
    }
    #pragma unroll
    for (int nt = 0; nt < 4; ++nt) {
      int hd = nt * 16 + lm;
      v8bf lo = *(const v8bf*)(VT + hd * 32 + 16 * hw);
      v8bf hi = *(const v8bf*)(VT + hd * 32 + 16 * hw + 8);
      v16bf bv;
      #pragma unroll
      for (int e = 0; e < 8; ++e) { bv[e] = lo[e]; bv[8 + e] = hi[e]; }
      o[nt] = wmma_bf(ap, bv, o[nt]);
    }
    __syncthreads();
  }

  // context out (bf16), row i, col h*64 + nt*16 + lm
  #pragma unroll
  for (int nt = 0; nt < 4; ++nt) {
    #pragma unroll
    for (int r = 0; r < 8; ++r) {
      int i   = q0 + r + 8 * hw;
      int col = h * HEAD_DIM + nt * 16 + lm;
      C[(size_t)(b * SEQ + i) * D_MODEL + col] = (bf16_t)(o[nt][r] / l_run[r]);
    }
  }
}

// ---------------------------------------------------------------- launch
extern "C" void kernel_launch(void* const* d_in, const int* in_sizes, int n_in,
                              void* d_out, int out_size, void* d_ws, size_t ws_size,
                              hipStream_t stream) {
  const float* q     = (const float*)d_in[0];
  const float* k     = (const float*)d_in[1];
  const float* v     = (const float*)d_in[2];
  const float* Wq    = (const float*)d_in[3];
  const float* biasq = (const float*)d_in[4];
  const float* Wk    = (const float*)d_in[5];
  const float* biask = (const float*)d_in[6];
  const float* Wv    = (const float*)d_in[7];
  const float* biasv = (const float*)d_in[8];
  const float* Wo    = (const float*)d_in[9];
  const float* biaso = (const float*)d_in[10];
  float* out = (float*)d_out;

  const size_t EL = (size_t)BATCH * SEQ * D_MODEL;  // 4,194,304
  const size_t WE = (size_t)D_MODEL * D_MODEL;      // 1,048,576

  bf16_t* p   = (bf16_t*)d_ws;
  bf16_t* Xq  = p;          bf16_t* Xk  = Xq + EL;  bf16_t* Xv  = Xk + EL;
  bf16_t* Wqb = Xv + EL;    bf16_t* Wkb = Wqb + WE; bf16_t* Wvb = Wkb + WE;
  bf16_t* Wob = Wvb + WE;
  bf16_t* Qb  = Wob + WE;   bf16_t* Kb  = Qb + EL;  bf16_t* Vb  = Kb + EL;
  bf16_t* Cb  = Vb + EL;    // total ~67 MB of ws

  dim3 blk(256);
  cvt_f32_bf16<<<2048, blk, 0, stream>>>(q,  Xq,  (int)EL);
  cvt_f32_bf16<<<2048, blk, 0, stream>>>(k,  Xk,  (int)EL);
  cvt_f32_bf16<<<2048, blk, 0, stream>>>(v,  Xv,  (int)EL);
  cvt_f32_bf16<<<512,  blk, 0, stream>>>(Wq, Wqb, (int)WE);
  cvt_f32_bf16<<<512,  blk, 0, stream>>>(Wk, Wkb, (int)WE);
  cvt_f32_bf16<<<512,  blk, 0, stream>>>(Wv, Wvb, (int)WE);
  cvt_f32_bf16<<<512,  blk, 0, stream>>>(Wo, Wob, (int)WE);

  const int M = BATCH * SEQ, N = D_MODEL, K = D_MODEL;
  const int gemm_blocks = (M / 16) * (N / 64) / 8;  // 512 blocks x 8 waves
  gemm_xwt<false><<<gemm_blocks, blk, 0, stream>>>(Xq, Wqb, biasq, Qb, M, N, K);
  gemm_xwt<false><<<gemm_blocks, blk, 0, stream>>>(Xk, Wkb, biask, Kb, M, N, K);
  gemm_xwt<false><<<gemm_blocks, blk, 0, stream>>>(Xv, Wvb, biasv, Vb, M, N, K);

  const int attn_blocks = (BATCH * N_HEADS * (SEQ / 16)) / 8;  // 512
  attn_window<<<attn_blocks, blk, 0, stream>>>(Qb, Kb, Vb, Cb);

  gemm_xwt<true><<<gemm_blocks, blk, 0, stream>>>(Cb, Wob, biaso, out, M, N, K);
}
